// AttDual_24326694765174
// MI455X (gfx1250) — compile-verified, matrix operand order checked by reading, not run on hardware
//
#include <hip/hip_runtime.h>

#define D      1024
#define NROWS  50000
#define CLS    7

typedef __attribute__((ext_vector_type(16))) __bf16 v16bf;
typedef __attribute__((ext_vector_type(8)))  float  v8f;
typedef unsigned short u16;

union Frag {
  v16bf v;
  uint4 q[2];
  u16   u[16];
};

__device__ __forceinline__ u16 f2bf(float x) {
  unsigned int b = __float_as_uint(x);
  b += 0x7FFFu + ((b >> 16) & 1u);          // round-to-nearest-even
  return (u16)(b >> 16);
}
__device__ __forceinline__ float bf2f(u16 h) {
  return __uint_as_float(((unsigned int)h) << 16);
}
__device__ __forceinline__ float gelu_exact(float x) {
  return 0.5f * x * (1.0f + erff(x * 0.70710678118654752440f));
}

// ---- WMMA fragment loads (CDNA5 16-bit 16x32 layout) -----------------------
// lane<16  : halves 0..7 = K 0..7,   halves 8..15 = K 16..23 (of 32-chunk)
// lane>=16 : halves 0..7 = K 8..15,  halves 8..15 = K 24..31
// p16 points at (row base + k0 + klo); klo = (lane<16)?0:8.
__device__ __forceinline__ v16bf load_bf_frag(const u16* p16) {
  Frag f;
  const uint4* p = reinterpret_cast<const uint4*>(p16);
  f.q[0] = p[0];   // 8 halves at +0
  f.q[1] = p[2];   // 8 halves at +16 elements (=+32 bytes)
  return f.v;
}
__device__ __forceinline__ v16bf load_f32_frag(const float* p) {
  Frag f;
#pragma unroll
  for (int i = 0; i < 8; ++i) f.u[i] = f2bf(p[i]);
#pragma unroll
  for (int i = 0; i < 8; ++i) f.u[8 + i] = f2bf(p[16 + i]);
  return f.v;
}

// ---- fused Linear(+bias) -> LayerNorm -> GELU chain kernel -----------------
// One block = 32 rows x 1024 cols (two 16-row WMMA fragments share each B tile
// load: 20 b128 loads per 16 WMMAs).  8 waves, wave owns 128 cols (8 tiles).
// Row clamp + store guard handle Mtotal not divisible by 32.
// Optionally writes bf16 output, f32 output, and/or fused logits vs qmaxF.
template <bool INF32>
__global__ __launch_bounds__(256)
void chain_kernel(const float* __restrict__ Xf, const u16* __restrict__ Xb,
                  const u16* __restrict__ W,
                  const float* __restrict__ bias, const float* __restrict__ g,
                  const float* __restrict__ beta,
                  u16* __restrict__ Ybf, float* __restrict__ Yf32,
                  const float* __restrict__ qmaxF, float* __restrict__ logits,
                  int Mtotal) {
  __shared__ u16   ybuf[16 * D];     // per-fragment stash for logit fusion
  __shared__ float ps[32 * 8];
  __shared__ float pss[32 * 8];
  __shared__ float s_mu[32];
  __shared__ float s_rs[32];

  const int  tid     = threadIdx.x;
  const int  lane    = tid & 31;
  const int  wave    = tid >> 5;
  const int  m       = lane & 15;
  const int  klo     = (lane < 16) ? 0 : 8;
  const int  rowAdd  = (lane < 16) ? 0 : 8;
  const int  colBase = wave * 128;
  const long rowBase = (long)blockIdx.x * 32;

  // clamped source rows for the two 16-row fragments (keeps loads in-bounds)
  long r0 = rowBase + m;       if (r0 > (long)Mtotal - 1) r0 = Mtotal - 1;
  long r1 = rowBase + 16 + m;  if (r1 > (long)Mtotal - 1) r1 = Mtotal - 1;

  const float* xf0 = nullptr; const float* xf1 = nullptr;
  const u16*   xb0 = nullptr; const u16*   xb1 = nullptr;
  if (INF32) { xf0 = Xf + r0 * D + klo; xf1 = Xf + r1 * D + klo; }
  else       { xb0 = Xb + r0 * D + klo; xb1 = Xb + r1 * D + klo; }

  const u16* wrow[8];
#pragma unroll
  for (int t = 0; t < 8; ++t)
    wrow[t] = W + (long)(colBase + t * 16 + m) * D + klo;

  v8f acc[2][8] = {};
  for (int k0 = 0; k0 < D; k0 += 32) {
    v16bf a0 = INF32 ? load_f32_frag(xf0 + k0) : load_bf_frag(xb0 + k0);
    v16bf a1 = INF32 ? load_f32_frag(xf1 + k0) : load_bf_frag(xb1 + k0);
#pragma unroll
    for (int t = 0; t < 8; ++t) {
      v16bf b = load_bf_frag(wrow[t] + k0);
      acc[0][t] = __builtin_amdgcn_wmma_f32_16x16x32_bf16(
          false, a0, false, b, (short)0, acc[0][t], false, false);
      acc[1][t] = __builtin_amdgcn_wmma_f32_16x16x32_bf16(
          false, a1, false, b, (short)0, acc[1][t], false, false);
    }
  }

  // bias add + per-row partial sums
  float gv[8], bv[8];
#pragma unroll
  for (int t = 0; t < 8; ++t) {
    int col = colBase + t * 16 + m;
    float bc = bias[col];
    gv[t] = g[col];
    bv[t] = beta[col];
#pragma unroll
    for (int f = 0; f < 2; ++f)
#pragma unroll
      for (int r = 0; r < 8; ++r) acc[f][t][r] += bc;
  }
  float s8[2][8], q8[2][8];
#pragma unroll
  for (int f = 0; f < 2; ++f)
#pragma unroll
    for (int r = 0; r < 8; ++r) { s8[f][r] = 0.f; q8[f][r] = 0.f; }
#pragma unroll
  for (int f = 0; f < 2; ++f)
#pragma unroll
    for (int t = 0; t < 8; ++t)
#pragma unroll
      for (int r = 0; r < 8; ++r) {
        float v = acc[f][t][r];
        s8[f][r] += v;
        q8[f][r] += v * v;
      }
  // reduce across the 16 lanes of each half-wave (fixed order: deterministic)
#pragma unroll
  for (int f = 0; f < 2; ++f)
#pragma unroll
    for (int r = 0; r < 8; ++r)
      for (int off = 8; off > 0; off >>= 1) {
        s8[f][r] += __shfl_xor(s8[f][r], off, 16);
        q8[f][r] += __shfl_xor(q8[f][r], off, 16);
      }
  if (m == 0) {
#pragma unroll
    for (int f = 0; f < 2; ++f)
#pragma unroll
      for (int r = 0; r < 8; ++r) {
        ps [(f * 16 + rowAdd + r) * 8 + wave] = s8[f][r];
        pss[(f * 16 + rowAdd + r) * 8 + wave] = q8[f][r];
      }
  }
  __syncthreads();
  if (tid < 32) {
    float s = 0.f, q = 0.f;
    for (int w = 0; w < 8; ++w) { s += ps[tid * 8 + w]; q += pss[tid * 8 + w]; }
    float mean = s * (1.0f / D);
    float var  = q * (1.0f / D) - mean * mean;
    s_mu[tid] = mean;
    s_rs[tid] = 1.0f / sqrtf(var + 1e-5f);
  }
  __syncthreads();

  // LN + GELU epilogue, one 16-row fragment at a time (shares 32KB LDS stash)
  for (int f = 0; f < 2; ++f) {
#pragma unroll
    for (int t = 0; t < 8; ++t) {
      int col = colBase + t * 16 + m;
#pragma unroll
      for (int r = 0; r < 8; ++r) {
        int   row  = f * 16 + rowAdd + r;
        long  gRow = rowBase + row;
        float y = (acc[f][t][r] - s_mu[row]) * s_rs[row] * gv[t] + bv[t];
        y = gelu_exact(y);
        u16 yb = f2bf(y);
        if (gRow < Mtotal) {
          if (Ybf)  Ybf [gRow * D + col] = yb;
          if (Yf32) Yf32[gRow * D + col] = y;
        }
        if (qmaxF) ybuf[(rowAdd + r) * D + col] = yb;
      }
    }
    if (qmaxF) {  // fused logits: (Y @ qmax^T) / sqrt(d) for this fragment
      __syncthreads();
      if (tid < 16 * CLS) {
        int  r = tid / CLS, cls = tid % CLS;
        long gRow = rowBase + f * 16 + r;
        if (gRow < Mtotal) {
          const float* qm = qmaxF + cls * D;
          float s = 0.f;
          for (int k = 0; k < D; ++k) s += bf2f(ybuf[r * D + k]) * qm[k];
          logits[gRow * CLS + cls] = s * 0.03125f;
        }
      }
      __syncthreads();
    }
  }
}

// ---- small helper kernels --------------------------------------------------
__global__ void f32_to_bf16_kernel(const float* __restrict__ src,
                                   u16* __restrict__ dst, int n) {
  int i = blockIdx.x * blockDim.x + threadIdx.x;
  if (i < n) dst[i] = f2bf(src[i]);
}

__global__ __launch_bounds__(256)
void argmax_part_kernel(const float* __restrict__ cmat, float* __restrict__ pval,
                        int* __restrict__ pidx, int n) {
  __shared__ float sv[256];
  __shared__ int   si[256];
  const int t  = threadIdx.x;
  const int CH = (n + 255) / 256;
  int n0 = blockIdx.x * CH;
  int n1 = n0 + CH; if (n1 > n) n1 = n;
  float bvv[CLS]; int bii[CLS];
#pragma unroll
  for (int c = 0; c < CLS; ++c) { bvv[c] = -3.4e38f; bii[c] = 0x7FFFFFFF; }
  for (int nn = n0 + t; nn < n1; nn += 256) {
#pragma unroll
    for (int c = 0; c < CLS; ++c) {
      float v = cmat[(long)nn * CLS + c];
      if (v > bvv[c]) { bvv[c] = v; bii[c] = nn; }
    }
  }
  for (int c = 0; c < CLS; ++c) {
    sv[t] = bvv[c]; si[t] = bii[c];
    __syncthreads();
    for (int off = 128; off > 0; off >>= 1) {
      if (t < off) {
        float v1 = sv[t + off]; int i1 = si[t + off];
        if (v1 > sv[t] || (v1 == sv[t] && i1 < si[t])) { sv[t] = v1; si[t] = i1; }
      }
      __syncthreads();
    }
    if (t == 0) { pval[blockIdx.x * CLS + c] = sv[0]; pidx[blockIdx.x * CLS + c] = si[0]; }
    __syncthreads();
  }
}

__global__ __launch_bounds__(256)
void argmax_fin_gather_kernel(const float* __restrict__ pval,
                              const int* __restrict__ pidx,
                              const u16* __restrict__ Kbf,
                              u16* __restrict__ mfeats, int nblocks) {
  __shared__ int top[CLS];
  int t = threadIdx.x;
  if (t < CLS) {
    float bv = -3.4e38f; int bi = 0x7FFFFFFF;
    for (int b = 0; b < nblocks; ++b) {
      float v = pval[b * CLS + t]; int i = pidx[b * CLS + t];
      if (v > bv || (v == bv && i < bi)) { bv = v; bi = i; }
    }
    top[t] = bi;
  }
  __syncthreads();
  for (int idx = t; idx < 16 * D; idx += 256) {
    int row = idx >> 10, col = idx & (D - 1);
    mfeats[idx] = (row < CLS) ? Kbf[(long)top[row] * D + col] : (u16)0;
  }
}

__global__ __launch_bounds__(256)
void smax_part_kernel(const float* __restrict__ logits,
                      float* __restrict__ pmax, int n) {
  __shared__ float sv[256];
  const int t  = threadIdx.x;
  const int CH = (n + 255) / 256;
  int n0 = blockIdx.x * CH;
  int n1 = n0 + CH; if (n1 > n) n1 = n;
  float mv[CLS];
#pragma unroll
  for (int c = 0; c < CLS; ++c) mv[c] = -3.4e38f;
  for (int nn = n0 + t; nn < n1; nn += 256)
#pragma unroll
    for (int c = 0; c < CLS; ++c) mv[c] = fmaxf(mv[c], logits[(long)nn * CLS + c]);
  for (int c = 0; c < CLS; ++c) {
    sv[t] = mv[c];
    __syncthreads();
    for (int off = 128; off > 0; off >>= 1) {
      if (t < off) sv[t] = fmaxf(sv[t], sv[t + off]);
      __syncthreads();
    }
    if (t == 0) pmax[blockIdx.x * CLS + c] = sv[0];
    __syncthreads();
  }
}

__global__ void smax_fin_kernel(const float* __restrict__ pmax,
                                float* __restrict__ cmax, int nb) {
  int t = threadIdx.x;
  if (t < CLS) {
    float m = -3.4e38f;
    for (int b = 0; b < nb; ++b) m = fmaxf(m, pmax[b * CLS + t]);
    cmax[t] = m;
  }
}

__global__ __launch_bounds__(256)
void ssum_part_kernel(const float* __restrict__ logits,
                      const float* __restrict__ cmax,
                      float* __restrict__ psum, int n) {
  __shared__ float sv[256];
  const int t  = threadIdx.x;
  const int CH = (n + 255) / 256;
  int n0 = blockIdx.x * CH;
  int n1 = n0 + CH; if (n1 > n) n1 = n;
  float cm[CLS], sm[CLS];
#pragma unroll
  for (int c = 0; c < CLS; ++c) { cm[c] = cmax[c]; sm[c] = 0.f; }
  for (int nn = n0 + t; nn < n1; nn += 256)
#pragma unroll
    for (int c = 0; c < CLS; ++c)
      sm[c] += expf(logits[(long)nn * CLS + c] - cm[c]);
  for (int c = 0; c < CLS; ++c) {
    sv[t] = sm[c];
    __syncthreads();
    for (int off = 128; off > 0; off >>= 1) {
      if (t < off) sv[t] += sv[t + off];
      __syncthreads();
    }
    if (t == 0) psum[blockIdx.x * CLS + c] = sv[0];
    __syncthreads();
  }
}

__global__ void ssum_fin_kernel(const float* __restrict__ psum,
                                float* __restrict__ csum, int nb) {
  int t = threadIdx.x;
  if (t < CLS) {
    float s = 0.f;
    for (int b = 0; b < nb; ++b) s += psum[b * CLS + t];
    csum[t] = s;
  }
}

__global__ void write_A_kernel(const float* __restrict__ logits,
                               const float* __restrict__ cmax,
                               const float* __restrict__ csum,
                               float* __restrict__ A, int total) {
  int i = blockIdx.x * 256 + threadIdx.x;
  if (i < total) {
    int c = i % CLS;
    A[i] = expf(logits[i] - cmax[c]) / csum[c];
  }
}

__global__ __launch_bounds__(256)
void b_part_kernel(const float* __restrict__ A, const u16* __restrict__ Vbf,
                   float* __restrict__ Bpart, int chunk) {
  const int t  = threadIdx.x;
  long n0 = (long)blockIdx.x * chunk, n1 = n0 + chunk;
  float acc[CLS][4];
#pragma unroll
  for (int c = 0; c < CLS; ++c)
#pragma unroll
    for (int j = 0; j < 4; ++j) acc[c][j] = 0.f;
  for (long nn = n0; nn < n1; ++nn) {
    float a[CLS];
#pragma unroll
    for (int c = 0; c < CLS; ++c) a[c] = A[nn * CLS + c];
    const u16* vr = Vbf + nn * D;
#pragma unroll
    for (int j = 0; j < 4; ++j) {
      float v = bf2f(vr[t + j * 256]);
#pragma unroll
      for (int c = 0; c < CLS; ++c) acc[c][j] += a[c] * v;
    }
  }
#pragma unroll
  for (int c = 0; c < CLS; ++c)
#pragma unroll
    for (int j = 0; j < 4; ++j)
      Bpart[(long)blockIdx.x * CLS * D + c * D + t + j * 256] = acc[c][j];
}

__global__ __launch_bounds__(256)
void b_fin_c_kernel(const float* __restrict__ Bpart,
                    const float* __restrict__ head_w,
                    const float* __restrict__ head_b,
                    float* __restrict__ outB, float* __restrict__ outC) {
  __shared__ float Bl[CLS * D];
  __shared__ float red[256];
  int t = threadIdx.x;
  for (int idx = t; idx < CLS * D; idx += 256) {
    float s = 0.f;
    for (int ch = 0; ch < 8; ++ch) s += Bpart[ch * CLS * D + idx];
    Bl[idx] = s;
    outB[idx] = s;
  }
  __syncthreads();
  for (int o = 0; o < CLS; ++o) {
    float s = 0.f;
    for (int idx = t; idx < CLS * D; idx += 256) s += Bl[idx] * head_w[o * CLS * D + idx];
    red[t] = s;
    __syncthreads();
    for (int off = 128; off > 0; off >>= 1) {
      if (t < off) red[t] += red[t + off];
      __syncthreads();
    }
    if (t == 0) outC[o] = red[0] + head_b[o];
    __syncthreads();
  }
}

// ---- host orchestration ----------------------------------------------------
extern "C" void kernel_launch(void* const* d_in, const int* in_sizes, int n_in,
                              void* d_out, int out_size, void* d_ws, size_t ws_size,
                              hipStream_t stream) {
  (void)in_sizes; (void)n_in; (void)out_size; (void)ws_size;

  const float* features   = (const float*)d_in[0];
  const float* cmat       = (const float*)d_in[1];
  const float* key_w      = (const float*)d_in[2];
  const float* key_b      = (const float*)d_in[3];
  const float* key_g      = (const float*)d_in[4];
  const float* key_beta   = (const float*)d_in[5];
  const float* query_w    = (const float*)d_in[6];
  const float* query_b    = (const float*)d_in[7];
  const float* query_g    = (const float*)d_in[8];
  const float* query_beta = (const float*)d_in[9];
  const float* value_w    = (const float*)d_in[10];
  const float* value_b    = (const float*)d_in[11];
  const float* value_g    = (const float*)d_in[12];
  const float* value_beta = (const float*)d_in[13];
  const float* head_w     = (const float*)d_in[14];
  const float* head_b     = (const float*)d_in[15];

  float* out  = (float*)d_out;
  float* outC = out;                                 // [7]
  float* outA = out + CLS;                           // [N,7]
  float* outB = out + CLS + (long)NROWS * CLS;       // [7,1024]

  unsigned char* w8 = (unsigned char*)d_ws;
  size_t cur = 0;
  auto take = [&](size_t bytes) -> void* {
    cur = (cur + 255) & ~(size_t)255;
    void* p = w8 + cur;
    cur += bytes;
    return p;
  };
  u16*   wbf_k  = (u16*)  take((size_t)D * D * 2);
  u16*   wbf_q  = (u16*)  take((size_t)D * D * 2);
  u16*   wbf_v  = (u16*)  take((size_t)D * D * 2);
  u16*   Kbf    = (u16*)  take((size_t)NROWS * D * 2);
  u16*   Vbf    = (u16*)  take((size_t)NROWS * D * 2);
  u16*   mfeats = (u16*)  take((size_t)16 * D * 2);
  float* qmaxF  = (float*)take((size_t)16 * D * 4);
  float* logits = (float*)take((size_t)NROWS * CLS * 4);
  float* pval   = (float*)take((size_t)256 * CLS * 4);
  int*   pidx   = (int*)  take((size_t)256 * CLS * 4);
  float* pmax   = (float*)take((size_t)256 * CLS * 4);
  float* psum   = (float*)take((size_t)256 * CLS * 4);
  float* cmax   = (float*)take((size_t)CLS * 4);
  float* csum   = (float*)take((size_t)CLS * 4);
  float* Bpart  = (float*)take((size_t)8 * CLS * D * 4);

  dim3 blk(256);
  const int wgrid = (D * D + 255) / 256;
  f32_to_bf16_kernel<<<wgrid, blk, 0, stream>>>(key_w,   wbf_k, D * D);
  f32_to_bf16_kernel<<<wgrid, blk, 0, stream>>>(query_w, wbf_q, D * D);
  f32_to_bf16_kernel<<<wgrid, blk, 0, stream>>>(value_w, wbf_v, D * D);

  const int rb = (NROWS + 31) / 32;  // 1563 row blocks of 32

  // K = chain(features, key)
  chain_kernel<true><<<rb, blk, 0, stream>>>(features, nullptr, wbf_k,
      key_b, key_g, key_beta, Kbf, nullptr, nullptr, nullptr, NROWS);

  // top_idx = argmax(c, axis=0); m_feats = K[top_idx] (padded to 16 rows)
  argmax_part_kernel<<<256, blk, 0, stream>>>(cmat, pval, pidx, NROWS);
  argmax_fin_gather_kernel<<<1, blk, 0, stream>>>(pval, pidx, Kbf, mfeats, 256);

  // q_max = chain(m_feats, query), f32 output (16 padded rows, 7 valid)
  chain_kernel<false><<<1, blk, 0, stream>>>(nullptr, mfeats, wbf_q,
      query_b, query_g, query_beta, nullptr, qmaxF, nullptr, nullptr, 16);

  // V = chain(K, value)
  chain_kernel<false><<<rb, blk, 0, stream>>>(nullptr, Kbf, wbf_v,
      value_b, value_g, value_beta, Vbf, nullptr, nullptr, nullptr, NROWS);

  // Q = chain(K, query), fused with logits = Q @ q_max^T / 32 (Q never stored)
  chain_kernel<false><<<rb, blk, 0, stream>>>(nullptr, Kbf, wbf_q,
      query_b, query_g, query_beta, nullptr, nullptr, qmaxF, logits, NROWS);

  // softmax over axis 0
  smax_part_kernel<<<256, blk, 0, stream>>>(logits, pmax, NROWS);
  smax_fin_kernel<<<1, blk, 0, stream>>>(pmax, cmax, 256);
  ssum_part_kernel<<<256, blk, 0, stream>>>(logits, cmax, psum, NROWS);
  ssum_fin_kernel<<<1, blk, 0, stream>>>(psum, csum, 256);
  const int agrid = (NROWS * CLS + 255) / 256;
  write_A_kernel<<<agrid, blk, 0, stream>>>(logits, cmax, csum, outA, NROWS * CLS);

  // B = A^T @ V ; C = einsum('ik,oik->o', B, head_w) + head_b
  b_part_kernel<<<8, blk, 0, stream>>>(outA, Vbf, Bpart, NROWS / 8);
  b_fin_c_kernel<<<1, blk, 0, stream>>>(Bpart, head_w, head_b, outB, outC);
}
